// Decoder_44135083933715
// MI455X (gfx1250) — matrix operand order repaired and checked by reading, not compiled, not run on hardware
//
#include <hip/hip_runtime.h>
#include <hip/hip_bf16.h>

typedef _Float16 h16;
typedef __attribute__((ext_vector_type(16))) _Float16 v16h;
typedef __attribute__((ext_vector_type(8)))  _Float16 h8;
typedef __attribute__((ext_vector_type(4)))  _Float16 hv4;
typedef __attribute__((ext_vector_type(8)))  float    v8f;
typedef __attribute__((ext_vector_type(4)))  float    f4;

#define TILE     16
#define EL       48      // ENC_LEN
#define EH       128     // ENC_HID
#define DD       32      // DEC_DIM
#define TDEC     12
#define BTOT     16384
#define NTHREADS 128

// ---- LDS layout (bytes, all offsets 16B aligned) ----
constexpr int OFF_ENC16 = 0;                              // [16][48][128] f16
constexpr int OFF_CATXH = OFF_ENC16 + TILE*EL*EH*2;       // [16][64] f16 (x:0-3, h:4-35, pad->0)
constexpr int OFF_HF16  = OFF_CATXH + TILE*64*2;          // [16][32] f16
constexpr int OFF_WHH16 = OFF_HF16  + TILE*32*2;          // [96][32] f16
constexpr int OFF_AW16  = OFF_WHH16 + 96*32*2;            // [48][64] f16 (attn_W padded)
constexpr int OFF_HF32  = OFF_AW16  + EL*64*2;            // [16][32] f32
constexpr int OFF_XF32  = OFF_HF32  + TILE*32*4;          // [16][4]  f32
constexpr int OFF_LOG   = OFF_XF32  + TILE*4*4;           // [16][48] f32 (logits -> aw)
constexpr int OFF_X2P   = OFF_LOG   + TILE*EL*4;          // [16][8][4] f32
constexpr int OFF_X2    = OFF_X2P   + TILE*8*4*4;         // [16][4]  f32
constexpr int OFF_GI    = OFF_X2    + TILE*4*4;           // [16][96] f32
constexpr int OFF_GH    = OFF_GI    + TILE*96*4;          // [16][96] f32
constexpr int OFF_OUTP  = OFF_GH    + TILE*96*4;          // [16][8]  f32
constexpr int OFF_ATTNB = OFF_OUTP  + TILE*8*4;           // [48] f32
constexpr int OFF_COMBW = OFF_ATTNB + 48*4;               // [4][132] f32
constexpr int OFF_COMBB = OFF_COMBW + 4*132*4;            // [4] f32
constexpr int OFF_WIH   = OFF_COMBB + 16;                 // [96][4] f32
constexpr int OFF_BIH   = OFF_WIH   + 96*4*4;             // [96] f32
constexpr int OFF_BHH   = OFF_BIH   + 96*4;               // [96] f32
constexpr int OFF_FCW   = OFF_BHH   + 96*4;               // [32] f32
constexpr int OFF_FCB   = OFF_FCW   + 32*4;               // [1] f32
constexpr int SMEM_TOTAL= OFF_FCB   + 16;                 // ~232 KB (< 320 KB WGP LDS)

__device__ inline float nan2num(float v) {
    if (v != v) return 0.0f;
    if (v >  3.402823466e38f) return  3.402823466e38f;
    if (v < -3.402823466e38f) return -3.402823466e38f;
    return v;
}

__device__ inline v16h pack16(h8 lo, h8 hi) {
    v16h r;
#pragma unroll
    for (int i = 0; i < 8; ++i) { r[i] = lo[i]; r[8 + i] = hi[i]; }
    return r;
}

__global__ void __launch_bounds__(NTHREADS)
decoder_attn_gru_kernel(const float* __restrict__ y,
                        const float* __restrict__ enc,
                        const float* __restrict__ hidden,
                        const float* __restrict__ attn_W,
                        const float* __restrict__ attn_b,
                        const float* __restrict__ comb_W,
                        const float* __restrict__ comb_b,
                        const float* __restrict__ w_ih,
                        const float* __restrict__ w_hh,
                        const float* __restrict__ b_ih,
                        const float* __restrict__ b_hh,
                        const float* __restrict__ fc_W,
                        const float* __restrict__ fc_b,
                        float* __restrict__ out)
{
    extern __shared__ char smem[];
    h16*   enc16   = (h16*)  (smem + OFF_ENC16);
    h16*   catxh   = (h16*)  (smem + OFF_CATXH);
    h16*   hfh     = (h16*)  (smem + OFF_HF16);
    h16*   whh16   = (h16*)  (smem + OFF_WHH16);
    h16*   aW16    = (h16*)  (smem + OFF_AW16);
    float* hf32    = (float*)(smem + OFF_HF32);
    float* xf32    = (float*)(smem + OFF_XF32);
    float* logits  = (float*)(smem + OFF_LOG);
    float* x2p     = (float*)(smem + OFF_X2P);
    float* x2s     = (float*)(smem + OFF_X2);
    float* gis     = (float*)(smem + OFF_GI);
    float* ghs     = (float*)(smem + OFF_GH);
    float* outp    = (float*)(smem + OFF_OUTP);
    float* s_attnb = (float*)(smem + OFF_ATTNB);
    float* s_combW = (float*)(smem + OFF_COMBW);
    float* s_combb = (float*)(smem + OFF_COMBB);
    float* s_wih   = (float*)(smem + OFF_WIH);
    float* s_bih   = (float*)(smem + OFF_BIH);
    float* s_bhh   = (float*)(smem + OFF_BHH);
    float* s_fcW   = (float*)(smem + OFF_FCW);
    float* s_fcb   = (float*)(smem + OFF_FCB);

    const int tid  = threadIdx.x;
    const int lane = tid & 31;
    const int wave = tid >> 5;
    const int b0   = blockIdx.x * TILE;

    // ---- stage weights (convert GEMM operands to f16 in LDS) ----
    for (int i = tid; i < EL*64; i += NTHREADS) {
        int e = i >> 6, k = i & 63;
        aW16[i] = (h16)((k < 36) ? attn_W[e*36 + k] : 0.0f);
    }
    for (int i = tid; i < TILE*64; i += NTHREADS) catxh[i] = (h16)0.0f; // clears K padding
    for (int i = tid; i < 96*32;   i += NTHREADS) whh16[i] = (h16)w_hh[i];
    for (int i = tid; i < 48;      i += NTHREADS) s_attnb[i] = attn_b[i];
    for (int i = tid; i < 4*132;   i += NTHREADS) s_combW[i] = comb_W[i];
    for (int i = tid; i < 4;       i += NTHREADS) s_combb[i] = comb_b[i];
    for (int i = tid; i < 96*4;    i += NTHREADS) s_wih[i]  = w_ih[i];
    for (int i = tid; i < 96;      i += NTHREADS) { s_bih[i] = b_ih[i]; s_bhh[i] = b_hh[i]; }
    for (int i = tid; i < 32;      i += NTHREADS) s_fcW[i]  = fc_W[i];
    if (tid == 0) s_fcb[0] = fc_b[0];
    __syncthreads();  // padding zeros visible before h/x writes into catxh

    // ---- initial state: h0, x0 = nan_to_num(y[:,0,:]) ----
    for (int i = tid; i < TILE*32; i += NTHREADS) {
        int b = i >> 5, d = i & 31;
        float v = hidden[(size_t)(b0 + b)*32 + d];
        hf32[i] = v; hfh[i] = (h16)v; catxh[b*64 + 4 + d] = (h16)v;
    }
    if (tid < TILE) {
        int b = tid;
#pragma unroll
        for (int i = 0; i < 4; ++i) {
            float v = nan2num(y[((size_t)(b0 + b)*13 + 0)*4 + i]);
            xf32[b*4 + i] = v; catxh[b*64 + i] = (h16)v;
        }
    }

    // ---- stage encoder tile into LDS once (402 MB total HBM -> ~17us floor) ----
    // enc global layout: (48, B, 128); LDS layout: [b][e][128] f16
    for (int idx = tid; idx < TILE*EL*(EH/4); idx += NTHREADS) {
        int h4i = idx & 31;           // float4 within a 128-row (coalesced per wave)
        int row = idx >> 5;
        int e = row % EL, b = row / EL;
        const f4* gp = (const f4*)(enc + ((size_t)e*BTOT + b0 + b)*EH);
        f4 v = gp[h4i];
        hv4 hv; hv[0] = (h16)v[0]; hv[1] = (h16)v[1]; hv[2] = (h16)v[2]; hv[3] = (h16)v[3];
        *(hv4*)&enc16[(b*EL + e)*EH + h4i*4] = hv;
    }
    __syncthreads();

    const int m  = lane & 15;         // WMMA row (A) / D row base
    const int hi = lane >> 4;         // half-wave selector per ISA operand layout
    const int n  = lane & 15;         // WMMA column (B/D)

    for (int t = 0; t < TDEC; ++t) {
        // ================= (A) attn logits: [x,h](16x36->64) @ attn_W^T -> 16x48 =====
        if (wave < 3) {               // uniform per-wave branch: EXEC stays all-1s
            const int e0 = wave * 16;
            v8f acc = {};
#pragma unroll
            for (int c = 0; c < 2; ++c) {   // K chunks 0..31, 32..63 (zero padded past 36)
                const h16* ar = &catxh[m*64 + c*32 + (hi ? 8 : 0)];
                v16h A = pack16(*(const h8*)ar, *(const h8*)(ar + 16));
                const h16* br = &aW16[(e0 + n)*64 + c*32 + (hi ? 8 : 0)];
                v16h Bm = pack16(*(const h8*)br, *(const h8*)(br + 16));
                acc = __builtin_amdgcn_wmma_f32_16x16x32_f16(false, A, false, Bm,
                                                             (short)0, acc, false, false);
            }
            float bias = s_attnb[e0 + n];
#pragma unroll
            for (int v = 0; v < 8; ++v)
                logits[(v + (hi ? 8 : 0))*EL + e0 + n] = acc[v] + bias;
        }
        __syncthreads();

        // ================= (B) softmax over 48 (one thread per sample) ==============
        if (tid < TILE) {
            float* lr = &logits[tid*EL];
            float mx = lr[0];
            for (int e = 1; e < EL; ++e) mx = fmaxf(mx, lr[e]);
            float s = 0.0f;
            for (int e = 0; e < EL; ++e) { float ev = __expf(lr[e] - mx); lr[e] = ev; s += ev; }
            float inv = 1.0f / s;
            for (int e = 0; e < EL; ++e) lr[e] *= inv;
        }
        __syncthreads();

        // ===== (C) ctx[b] = sum_e aw*enc (per-sample GEMV, from LDS) + x2 partials ===
        {
            const int b = tid >> 3, hg = tid & 7;   // 8 threads x 16 channels per sample
            float ctxv[16];
#pragma unroll
            for (int i = 0; i < 16; ++i) ctxv[i] = 0.0f;
            const h16*   eb  = &enc16[b*EL*EH + hg*16];
            const float* awr = &logits[b*EL];
            for (int e = 0; e < EL; ++e) {
                float a = awr[e];
                const h8* p = (const h8*)(eb + e*EH);
                h8 v0 = p[0], v1 = p[1];
#pragma unroll
                for (int i = 0; i < 8; ++i) {
                    ctxv[i]     += a * (float)v0[i];
                    ctxv[8 + i] += a * (float)v1[i];
                }
            }
#pragma unroll
            for (int j = 0; j < 4; ++j) {
                const float* wr = &s_combW[j*132 + hg*16];
                float s = 0.0f;
#pragma unroll
                for (int i = 0; i < 16; ++i) s += wr[i] * ctxv[i];
                if (hg == 0) {
                    const float* wx = &s_combW[j*132 + 128];
#pragma unroll
                    for (int i = 0; i < 4; ++i) s += wx[i] * xf32[b*4 + i];
                    s += s_combb[j];
                }
                x2p[(b*8 + hg)*4 + j] = s;
            }
        }
        __syncthreads();

        // ================= (D) reduce x2 partials ====================================
        if (tid < TILE) {
            int b = tid;
#pragma unroll
            for (int j = 0; j < 4; ++j) {
                float s = 0.0f;
                for (int g = 0; g < 8; ++g) s += x2p[(b*8 + g)*4 + j];
                x2s[b*4 + j] = s;
            }
        }
        __syncthreads();

        // ====== (E) gi = x2 @ w_ih^T (VALU, K=4)  +  gh = h @ w_hh^T (WMMA, K=32) ====
        {
            const int b = tid >> 3, hg = tid & 7;
            float xv0 = x2s[b*4+0], xv1 = x2s[b*4+1], xv2 = x2s[b*4+2], xv3 = x2s[b*4+3];
#pragma unroll
            for (int gg = 0; gg < 12; ++gg) {
                int g = hg*12 + gg;
                const float* wr = &s_wih[g*4];
                gis[b*96 + g] = s_bih[g] + xv0*wr[0] + xv1*wr[1] + xv2*wr[2] + xv3*wr[3];
            }
        }
        {
            const h16* ar = &hfh[m*32 + (hi ? 8 : 0)];
            v16h A = pack16(*(const h8*)ar, *(const h8*)(ar + 16));
            for (int nt = wave; nt < 6; nt += 4) {   // 6 N-tiles of 96 gates across 4 waves
                const int g0 = nt * 16;
                const h16* br = &whh16[(g0 + n)*32 + (hi ? 8 : 0)];
                v16h Bm = pack16(*(const h8*)br, *(const h8*)(br + 16));
                v8f acc = {};
                acc = __builtin_amdgcn_wmma_f32_16x16x32_f16(false, A, false, Bm,
                                                             (short)0, acc, false, false);
                float bias = s_bhh[g0 + n];
#pragma unroll
                for (int v = 0; v < 8; ++v)
                    ghs[(v + (hi ? 8 : 0))*96 + g0 + n] = acc[v] + bias;
            }
        }
        __syncthreads();

        // ================= (F) GRU gates + fc partial ================================
        {
            const int b = tid >> 3, hg = tid & 7;
            float op = 0.0f;
#pragma unroll
            for (int k = 0; k < 4; ++k) {
                int d = hg*4 + k;
                float r  = 1.0f / (1.0f + __expf(-(gis[b*96 +      d] + ghs[b*96 +      d])));
                float z  = 1.0f / (1.0f + __expf(-(gis[b*96 + 32 + d] + ghs[b*96 + 32 + d])));
                float nn = tanhf(gis[b*96 + 64 + d] + r * ghs[b*96 + 64 + d]);
                float hn = (1.0f - z) * nn + z * hf32[b*32 + d];
                hf32[b*32 + d] = hn;
                hfh [b*32 + d] = (h16)hn;
                catxh[b*64 + 4 + d] = (h16)hn;
                op += hn * s_fcW[d];
            }
            outp[b*8 + hg] = op;
        }
        __syncthreads();

        // ================= (G) output + next x =======================================
        if (tid < TILE) {
            int b = tid;
            float o = s_fcb[0];
            for (int g = 0; g < 8; ++g) o += outp[b*8 + g];
            out[(size_t)t*BTOT + b0 + b] = o;
            const float* yr = &y[((size_t)(b0 + b)*13 + (t + 1))*4];
            float a = yr[1], c = yr[2], d = yr[3];
            xf32[b*4+0] = a; xf32[b*4+1] = c; xf32[b*4+2] = d; xf32[b*4+3] = o;
            catxh[b*64+0] = (h16)a; catxh[b*64+1] = (h16)c;
            catxh[b*64+2] = (h16)d; catxh[b*64+3] = (h16)o;
        }
        __syncthreads();
    }
}

extern "C" void kernel_launch(void* const* d_in, const int* in_sizes, int n_in,
                              void* d_out, int out_size, void* d_ws, size_t ws_size,
                              hipStream_t stream)
{
    (void)in_sizes; (void)n_in; (void)out_size; (void)d_ws; (void)ws_size;
    const float* y      = (const float*)d_in[0];
    const float* enc    = (const float*)d_in[1];
    const float* hidden = (const float*)d_in[2];
    // d_in[3] = batch_ids (unused by reference)
    const float* attn_W = (const float*)d_in[4];
    const float* attn_b = (const float*)d_in[5];
    const float* comb_W = (const float*)d_in[6];
    const float* comb_b = (const float*)d_in[7];
    const float* w_ih   = (const float*)d_in[8];
    const float* w_hh   = (const float*)d_in[9];
    const float* b_ih   = (const float*)d_in[10];
    const float* b_hh   = (const float*)d_in[11];
    const float* fc_W   = (const float*)d_in[12];
    const float* fc_b   = (const float*)d_in[13];
    float* out = (float*)d_out;

    dim3 grid(BTOT / TILE), block(NTHREADS);
    hipLaunchKernelGGL(decoder_attn_gru_kernel, grid, block, SMEM_TOTAL, stream,
                       y, enc, hidden, attn_W, attn_b, comb_W, comb_b,
                       w_ih, w_hh, b_ih, b_hh, fc_W, fc_b, out);
}
